// Bert_CRF_22763326668899
// MI455X (gfx1250) — matrix operand order repaired and checked by reading, not compiled, not run on hardware
//
#include <hip/hip_runtime.h>
#include <hip/hip_bf16.h>
#include <math.h>

// ---------------- problem constants (match reference) ----------------
constexpr int Bc  = 32;
constexpr int Sc  = 256;
constexpr int Hc  = 768;
constexpr int NLc = 12;
constexpr int NHc = 12;
constexpr int FFc = 3072;
constexpr int Lc  = 9;
constexpr int DHc = Hc / NHc;          // 64
constexpr int Tc  = Bc * Sc;           // 8192 tokens
#define SCALE_ (0.125f)                // 1/sqrt(64)

// ---------------- WMMA types ----------------
typedef __attribute__((ext_vector_type(16))) __bf16 v16bf;
typedef __attribute__((ext_vector_type(8)))  float  v8f;
typedef __attribute__((ext_vector_type(4)))  int    v4i;

#define WMMA_BF16(a, b, c) \
  __builtin_amdgcn_wmma_f32_16x16x32_bf16(false, (a), false, (b), (short)0, (c), false, false)

// ---------------- async global->LDS staging (CDNA5) with fallback ----------------
#if defined(__HIP_DEVICE_COMPILE__) && __has_builtin(__builtin_amdgcn_global_load_async_to_lds_b128)
#define HAVE_ASYNC 1
#else
#define HAVE_ASYNC 0
#endif

#if HAVE_ASYNC
typedef __attribute__((address_space(1))) v4i GV4;
typedef __attribute__((address_space(3))) v4i LV4;
__device__ __forceinline__ void copy16(void* l, const void* g) {
  __builtin_amdgcn_global_load_async_to_lds_b128((GV4*)g, (LV4*)l, 0, 0);
}
#if __has_builtin(__builtin_amdgcn_s_wait_asynccnt)
#define ASYNC_WAIT(n) __builtin_amdgcn_s_wait_asynccnt(n)
#else
#define ASYNC_WAIT(n) asm volatile("s_wait_asynccnt %0" ::"n"(n))
#endif
#else
__device__ __forceinline__ void copy16(void* l, const void* g) {
  *(uint4*)l = *(const uint4*)g;
}
#define ASYNC_WAIT(n) ((void)0)
#endif

// A fragment (16x32 bf16, MxK) from LDS row-major [rows][lda] tile.
__device__ __forceinline__ v16bf frag_a_lds(const __bf16* p, int lda, int m0, int k0) {
  int lane = threadIdx.x & 31;
  int m    = m0 + (lane & 15);
  int half = lane >> 4;
  v16bf f;
  unsigned int* fd = (unsigned int*)&f;
#pragma unroll
  for (int v = 0; v < 8; ++v) {
    int kk = k0 + (v >> 2) * 16 + half * 8 + (v & 3) * 2;
    fd[v] = *(const unsigned int*)(p + (size_t)m * lda + kk);
  }
  return f;
}

// B fragment (32x16 bf16, KxN) gathered from LDS holding B^T row-major [n][ldb].
__device__ __forceinline__ v16bf frag_bT_lds(const __bf16* p, int ldb, int n0, int k0) {
  int lane = threadIdx.x & 31;
  int n    = n0 + (lane & 15);
  int half = lane >> 4;
  v16bf f;
  unsigned int* fd = (unsigned int*)&f;
#pragma unroll
  for (int v = 0; v < 8; ++v) {
    int kk = k0 + (v >> 2) * 16 + half * 8 + (v & 3) * 2;
    fd[v] = *(const unsigned int*)(p + (size_t)n * ldb + kk);
  }
  return f;
}

__device__ __forceinline__ float block_reduce_sum256(float v, float* red) {
  int tid = threadIdx.x;
  red[tid] = v;
  __syncthreads();
#pragma unroll
  for (int s = 128; s > 0; s >>= 1) {
    if (tid < s) red[tid] += red[tid + s];
    __syncthreads();
  }
  float r = red[0];
  __syncthreads();
  return r;
}

// ---------------- weight convert + transpose: W[K][N] f32 -> Wt[N][K] bf16 ----------------
__global__ __launch_bounds__(256) void bertcrf_cvt_t(
    const float* __restrict__ W, __bf16* __restrict__ Wt, int K, int N) {
  int idx = blockIdx.x * 256 + threadIdx.x;
  if (idx >= K * N) return;
  int kk = idx / N, nn = idx - kk * N;     // coalesced read of W
  Wt[(size_t)nn * K + kk] = (__bf16)W[idx];
}

// ---------------- embeddings + LayerNorm (writes fp32 h and bf16 copy) ----------------
__global__ __launch_bounds__(256) void bertcrf_embed_ln(
    const int* __restrict__ tok, const int* __restrict__ ttype,
    const float* __restrict__ we, const float* __restrict__ pe,
    const float* __restrict__ te, const float* __restrict__ g,
    const float* __restrict__ b, float* __restrict__ h, __bf16* __restrict__ hb) {
  __shared__ float red[256];
  int t   = blockIdx.x;
  int tid = threadIdx.x;
  int s   = t % Sc;
  int wid = tok[t];
  int ty  = ttype[t];
  float x[3];
  float sum = 0.f;
#pragma unroll
  for (int i = 0; i < 3; ++i) {
    int j = tid + i * 256;
    x[i]  = we[(size_t)wid * Hc + j] + pe[(size_t)s * Hc + j] + te[(size_t)ty * Hc + j];
    sum += x[i];
  }
  float mu = block_reduce_sum256(sum, red) * (1.0f / Hc);
  float sq = 0.f;
#pragma unroll
  for (int i = 0; i < 3; ++i) { float d = x[i] - mu; sq += d * d; }
  float var  = block_reduce_sum256(sq, red) * (1.0f / Hc);
  float rstd = rsqrtf(var + 1e-12f);
#pragma unroll
  for (int i = 0; i < 3; ++i) {
    int j = tid + i * 256;
    float o = (x[i] - mu) * rstd * g[j] + b[j];
    h[(size_t)t * Hc + j]  = o;
    hb[(size_t)t * Hc + j] = (__bf16)o;
  }
}

// ---------------- residual add + LayerNorm: h = LN(h + d); hb = bf16(h) ----------------
__global__ __launch_bounds__(256) void bertcrf_add_ln(
    float* __restrict__ h, const float* __restrict__ d,
    const float* __restrict__ g, const float* __restrict__ b,
    __bf16* __restrict__ hb) {
  __shared__ float red[256];
  int t   = blockIdx.x;
  int tid = threadIdx.x;
  float x[3];
  float sum = 0.f;
#pragma unroll
  for (int i = 0; i < 3; ++i) {
    int j = tid + i * 256;
    x[i]  = h[(size_t)t * Hc + j] + d[(size_t)t * Hc + j];
    sum += x[i];
  }
  float mu = block_reduce_sum256(sum, red) * (1.0f / Hc);
  float sq = 0.f;
#pragma unroll
  for (int i = 0; i < 3; ++i) { float dd = x[i] - mu; sq += dd * dd; }
  float var  = block_reduce_sum256(sq, red) * (1.0f / Hc);
  float rstd = rsqrtf(var + 1e-12f);
#pragma unroll
  for (int i = 0; i < 3; ++i) {
    int j = tid + i * 256;
    float o = (x[i] - mu) * rstd * g[j] + b[j];
    h[(size_t)t * Hc + j]  = o;
    hb[(size_t)t * Hc + j] = (__bf16)o;
  }
}

// ---------------- BF16 WMMA GEMM: out = epi(A[MxK] @ Wt^T + bias) ----------------
// A bf16 row-major [M][K]; Wt bf16 row-major [N][K] (pre-transposed weights).
// Block tile 128x128, K-step 32, async double-buffered staging, 8 waves (2x4).
template <bool GELU, bool OUTBF>
__global__ __launch_bounds__(256) void bertcrf_gemm(
    const __bf16* __restrict__ A, const __bf16* __restrict__ Wt,
    const float* __restrict__ bias, void* __restrict__ Cout,
    int M, int N, int K) {
  __shared__ __align__(16) __bf16 sA[2][128 * 32];
  __shared__ __align__(16) __bf16 sBT[2][128 * 32];

  int m0   = blockIdx.y * 128;
  int n0   = blockIdx.x * 128;
  int tid  = threadIdx.x;
  int w    = tid >> 5;
  int lane = tid & 31;
  int wr   = w >> 2;   // 0..1
  int wc   = w & 3;    // 0..3

  v8f acc[4][2];
#pragma unroll
  for (int i = 0; i < 4; ++i)
#pragma unroll
    for (int j = 0; j < 2; ++j)
#pragma unroll
      for (int r = 0; r < 8; ++r) acc[i][j][r] = 0.f;

  // stage one 128x32 A tile + one 128x32 Wt tile (4 x b128 per thread)
  auto stage = [&](int buf, int k0) {
#pragma unroll
    for (int i = 0; i < 2; ++i) {
      int chunk = i * 256 + tid;              // 512 chunks of 16B
      int r  = chunk >> 2;
      int c8 = (chunk & 3) * 8;
      copy16(&sA[buf][r * 32 + c8],  A  + (size_t)(m0 + r) * K + k0 + c8);
      copy16(&sBT[buf][r * 32 + c8], Wt + (size_t)(n0 + r) * K + k0 + c8);
    }
  };

  const int nIter = K >> 5;
  stage(0, 0);
  for (int it = 0; it < nIter; ++it) {
    int cur = it & 1;
    if (it + 1 < nIter) {
      stage(cur ^ 1, (it + 1) * 32);
      ASYNC_WAIT(4);
    } else {
      ASYNC_WAIT(0);
    }
    __syncthreads();

    v16bf af[4], bf[2];
#pragma unroll
    for (int i = 0; i < 4; ++i) af[i] = frag_a_lds(sA[cur], 32, wr * 64 + i * 16, 0);
#pragma unroll
    for (int j = 0; j < 2; ++j) bf[j] = frag_bT_lds(sBT[cur], 32, wc * 32 + j * 16, 0);
#pragma unroll
    for (int i = 0; i < 4; ++i)
#pragma unroll
      for (int j = 0; j < 2; ++j) acc[i][j] = WMMA_BF16(af[i], bf[j], acc[i][j]);
    __syncthreads();
  }

  int halfm = (lane >> 4) * 8;
  int nn    = lane & 15;
#pragma unroll
  for (int i = 0; i < 4; ++i)
#pragma unroll
    for (int j = 0; j < 2; ++j)
#pragma unroll
      for (int r = 0; r < 8; ++r) {
        int row = m0 + wr * 64 + i * 16 + halfm + r;
        int col = n0 + wc * 32 + j * 16 + nn;
        float v = acc[i][j][r] + bias[col];
        if (GELU) {
          float x = v;
          v = 0.5f * x * (1.0f + tanhf(0.7978845608f * (x + 0.044715f * x * x * x)));
        }
        if (OUTBF) ((__bf16*)Cout)[(size_t)row * N + col] = (__bf16)v;
        else       ((float*)Cout)[(size_t)row * N + col]  = v;
      }
}

// ---------------- fused attention (bf16 in/out): scores -> softmax -> @V ----------------
__global__ __launch_bounds__(256) void bertcrf_attn(
    const __bf16* __restrict__ q, const __bf16* __restrict__ k,
    const __bf16* __restrict__ v, const int* __restrict__ amask,
    __bf16* __restrict__ ctxb) {
  __shared__ __align__(16) __bf16 sQ[32 * 64];
  __shared__ __align__(16) __bf16 sKV[2][64 * 64];
  __shared__ __align__(16) float  sS[32 * 256];   // scores fp32, then probs bf16 in place

  int bh  = blockIdx.x;        // 0..B*NH-1
  int qt  = blockIdx.y;        // 0..S/32-1
  int b   = bh / NHc;
  int hh  = bh % NHc;
  int tid = threadIdx.x;
  int w   = tid >> 5;
  int lane = tid & 31;
  int mi = w >> 2;             // 0..1
  int ni = w & 3;              // 0..3

  // stage Q tile 32x64 bf16 (1 x b128 per thread)
  {
    int r  = tid >> 3;
    int c8 = (tid & 7) * 8;
    copy16(&sQ[r * 64 + c8], q + ((size_t)(b * Sc + qt * 32 + r)) * Hc + hh * DHc + c8);
  }
  auto stageK = [&](int buf, int kt) {
#pragma unroll
    for (int i = 0; i < 2; ++i) {
      int chunk = i * 256 + tid;             // 512 chunks of 16B
      int r  = chunk >> 3;
      int c8 = (chunk & 7) * 8;
      copy16(&sKV[buf][r * 64 + c8], k + ((size_t)(b * Sc + kt * 64 + r)) * Hc + hh * DHc + c8);
    }
  };

  // ---- phase 1: scores (double-buffered K tiles) ----
  stageK(0, 0);
  for (int kt = 0; kt < 4; ++kt) {
    int cur = kt & 1;
    if (kt < 3) {
      stageK(cur ^ 1, kt + 1);
      ASYNC_WAIT(2);
    } else {
      ASYNC_WAIT(0);
    }
    __syncthreads();

    v8f acc;
#pragma unroll
    for (int r = 0; r < 8; ++r) acc[r] = 0.f;
#pragma unroll
    for (int step = 0; step < 2; ++step) {
      v16bf af = frag_a_lds(sQ, 64, mi * 16, step * 32);
      v16bf bf = frag_bT_lds(sKV[cur], 64, ni * 16, step * 32);
      acc = WMMA_BF16(af, bf, acc);
    }
    int nn = lane & 15;
    int halfm = (lane >> 4) * 8;
    int key = kt * 64 + ni * 16 + nn;
    float mb = (1.0f - (float)amask[b * Sc + key]) * -1e9f;
#pragma unroll
    for (int r = 0; r < 8; ++r) {
      int srow = mi * 16 + halfm + r;
      sS[srow * 256 + key] = acc[r] * SCALE_ + mb;
    }
    __syncthreads();
  }

  // ---- phase 2: row softmax (wave handles 4 rows), probs bf16 in place ----
#pragma unroll
  for (int rr = 0; rr < 4; ++rr) {
    int r = w * 4 + rr;
    float vals[8];
#pragma unroll
    for (int i = 0; i < 8; ++i) vals[i] = sS[r * 256 + lane + i * 32];
    float mx = vals[0];
#pragma unroll
    for (int i = 1; i < 8; ++i) mx = fmaxf(mx, vals[i]);
#pragma unroll
    for (int off = 16; off > 0; off >>= 1) mx = fmaxf(mx, __shfl_xor(mx, off));
    float sum = 0.f;
#pragma unroll
    for (int i = 0; i < 8; ++i) { vals[i] = __expf(vals[i] - mx); sum += vals[i]; }
#pragma unroll
    for (int off = 16; off > 0; off >>= 1) sum += __shfl_xor(sum, off);
    float inv = 1.0f / sum;
    __bf16* prow = (__bf16*)(sS + (size_t)r * 256);   // row stride 512 bf16
#pragma unroll
    for (int i = 0; i < 8; ++i) prow[lane + i * 32] = (__bf16)(vals[i] * inv);
  }
  __syncthreads();

  // ---- phase 3: ctx = P @ V (V staged transposed with packed b32 stores) ----
  v8f acc2;
#pragma unroll
  for (int r = 0; r < 8; ++r) acc2[r] = 0.f;
  const __bf16* probs = (const __bf16*)sS;           // lda = 512 bf16
  for (int kt = 0; kt < 4; ++kt) {
#pragma unroll
    for (int i = 0; i < 8; ++i) {          // vT[d][key_local]: 2048 packed u32
      int idx = i * 256 + tid;
      int d  = idx & 63;
      int pr = idx >> 6;                   // key pair 0..31
      union { __bf16 h2[2]; unsigned u; } pk;
      pk.h2[0] = v[((size_t)(b * Sc + kt * 64 + pr * 2 + 0)) * Hc + hh * DHc + d];
      pk.h2[1] = v[((size_t)(b * Sc + kt * 64 + pr * 2 + 1)) * Hc + hh * DHc + d];
      ((unsigned*)sKV[0])[d * 32 + pr] = pk.u;
    }
    __syncthreads();
#pragma unroll
    for (int step = 0; step < 2; ++step) {
      v16bf af = frag_a_lds(probs, 512, mi * 16, kt * 64 + step * 32);
      v16bf bf = frag_bT_lds(sKV[0], 64, ni * 16, step * 32);
      acc2 = WMMA_BF16(af, bf, acc2);
    }
    __syncthreads();
  }
  int nn = lane & 15;
  int halfm = (lane >> 4) * 8;
#pragma unroll
  for (int r = 0; r < 8; ++r) {
    int srow = qt * 32 + mi * 16 + halfm + r;
    int d    = ni * 16 + nn;
    ctxb[((size_t)(b * Sc + srow)) * Hc + hh * DHc + d] = (__bf16)acc2[r];
  }
}

// ---------------- classifier: logits = log_softmax(h @ fc_w) ----------------
__global__ __launch_bounds__(256) void bertcrf_logits(
    const float* __restrict__ h, const float* __restrict__ fcw,
    float* __restrict__ logits) {
  __shared__ float red[256];
  __shared__ float tot[Lc];
  int row = blockIdx.x;
  int tid = threadIdx.x;
  for (int l = 0; l < Lc; ++l) {
    float s = 0.f;
#pragma unroll
    for (int i = 0; i < 3; ++i) {
      int j = tid + i * 256;
      s += h[(size_t)row * Hc + j] * fcw[(size_t)j * Lc + l];
    }
    float r = block_reduce_sum256(s, red);
    if (tid == 0) tot[l] = r;
  }
  __syncthreads();
  if (tid == 0) {
    float mx = tot[0];
    for (int l = 1; l < Lc; ++l) mx = fmaxf(mx, tot[l]);
    float se = 0.f;
    for (int l = 0; l < Lc; ++l) se += __expf(tot[l] - mx);
    float lse = __logf(se) + mx;
    for (int l = 0; l < Lc; ++l) logits[(size_t)row * Lc + l] = tot[l] - lse;
  }
}

__global__ void bertcrf_zero(float* o) { if (threadIdx.x == 0) o[0] = 0.f; }

// ---------------- CRF NLL: one wave32 per sequence; lanes 0..8 hold alpha ----------------
__global__ __launch_bounds__(32) void bertcrf_crf(
    const float* __restrict__ logits, const int* __restrict__ tags,
    const int* __restrict__ amask, const float* __restrict__ start,
    const float* __restrict__ endv, const float* __restrict__ trans,
    float* __restrict__ out) {
  int b    = blockIdx.x;
  int lane = threadIdx.x;
  const float NEG = -1e30f;

  int tag0 = tags[b * Sc + 0];
  float alpha = (lane < Lc) ? (start[lane] + logits[((size_t)(b * Sc)) * Lc + lane]) : NEG;
  float num = 0.f;
  int prev = tag0;
  if (lane == 0) num = start[tag0] + logits[((size_t)(b * Sc)) * Lc + tag0];

  for (int t = 1; t < Sc; ++t) {
    int m = amask[b * Sc + t];
    float av[Lc];
#pragma unroll
    for (int i = 0; i < Lc; ++i) av[i] = __shfl(alpha, i);
    if (lane < Lc) {
      float best = NEG;
#pragma unroll
      for (int i = 0; i < Lc; ++i) best = fmaxf(best, av[i] + trans[i * Lc + lane]);
      float sum = 0.f;
#pragma unroll
      for (int i = 0; i < Lc; ++i) sum += __expf(av[i] + trans[i * Lc + lane] - best);
      float newa = __logf(sum) + best + logits[((size_t)(b * Sc + t)) * Lc + lane];
      alpha = m ? newa : alpha;
    }
    if (lane == 0 && m) {
      int tg = tags[b * Sc + t];
      num += trans[prev * Lc + tg] + logits[((size_t)(b * Sc + t)) * Lc + tg];
      prev = tg;
    }
    prev = __shfl(prev, 0);
  }

  float aend = (lane < Lc) ? (alpha + endv[lane]) : NEG;
  float av[Lc];
#pragma unroll
  for (int i = 0; i < Lc; ++i) av[i] = __shfl(aend, i);
  if (lane == 0) {
    float best = NEG;
#pragma unroll
    for (int i = 0; i < Lc; ++i) best = fmaxf(best, av[i]);
    float sum = 0.f;
#pragma unroll
    for (int i = 0; i < Lc; ++i) sum += __expf(av[i] - best);
    float den = __logf(sum) + best;
    num += endv[prev];
    atomicAdd(out, den - num);   // loss = sum_b (den - num)
  }
}

// ---------------- host orchestration ----------------
extern "C" void kernel_launch(void* const* d_in, const int* in_sizes, int n_in,
                              void* d_out, int out_size, void* d_ws, size_t ws_size,
                              hipStream_t stream) {
  const int*   tok   = (const int*)d_in[0];
  const int*   ttype = (const int*)d_in[1];
  const int*   amask = (const int*)d_in[2];
  const int*   y     = (const int*)d_in[3];
  const float* we    = (const float*)d_in[4];
  const float* pe    = (const float*)d_in[5];
  const float* te    = (const float*)d_in[6];
  const float* eg    = (const float*)d_in[7];
  const float* eb    = (const float*)d_in[8];
  const float* Wq    = (const float*)d_in[9];
  const float* bq    = (const float*)d_in[10];
  const float* Wk    = (const float*)d_in[11];
  const float* bk    = (const float*)d_in[12];
  const float* Wv    = (const float*)d_in[13];
  const float* bv    = (const float*)d_in[14];
  const float* Wo    = (const float*)d_in[15];
  const float* bo    = (const float*)d_in[16];
  const float* ln1g  = (const float*)d_in[17];
  const float* ln1b  = (const float*)d_in[18];
  const float* W1    = (const float*)d_in[19];
  const float* fb1   = (const float*)d_in[20];
  const float* W2    = (const float*)d_in[21];
  const float* fb2   = (const float*)d_in[22];
  const float* ln2g  = (const float*)d_in[23];
  const float* ln2b  = (const float*)d_in[24];
  const float* fcw   = (const float*)d_in[25];
  const float* cs    = (const float*)d_in[26];
  const float* ce    = (const float*)d_in[27];
  const float* ct    = (const float*)d_in[28];

  char* base = (char*)d_ws;
  size_t off = 0;
  auto alloc = [&](size_t bytes) {
    void* p = base + off;
    off += (bytes + 255) & ~(size_t)255;
    return p;
  };
  float*  h    = (float*)alloc((size_t)Tc * Hc * 4);
  float*  tmp2 = (float*)alloc((size_t)Tc * Hc * 4);
  float*  lgts = (float*)alloc((size_t)Tc * Lc * 4);
  __bf16* hb   = (__bf16*)alloc((size_t)Tc * Hc * 2);
  __bf16* qb   = (__bf16*)alloc((size_t)Tc * Hc * 2);
  __bf16* kb   = (__bf16*)alloc((size_t)Tc * Hc * 2);
  __bf16* vb   = (__bf16*)alloc((size_t)Tc * Hc * 2);
  __bf16* ctxb = (__bf16*)alloc((size_t)Tc * Hc * 2);
  __bf16* ffb  = (__bf16*)alloc((size_t)Tc * FFc * 2);
  __bf16* wqT  = (__bf16*)alloc((size_t)NLc * Hc * Hc * 2);
  __bf16* wkT  = (__bf16*)alloc((size_t)NLc * Hc * Hc * 2);
  __bf16* wvT  = (__bf16*)alloc((size_t)NLc * Hc * Hc * 2);
  __bf16* woT  = (__bf16*)alloc((size_t)NLc * Hc * Hc * 2);
  __bf16* w1T  = (__bf16*)alloc((size_t)NLc * Hc * FFc * 2);
  __bf16* w2T  = (__bf16*)alloc((size_t)NLc * Hc * FFc * 2);

  // per-launch weight convert+transpose to bf16 [N][K]
  int gHH = (Hc * Hc + 255) / 256;
  int gHF = (Hc * FFc + 255) / 256;
  for (int l = 0; l < NLc; ++l) {
    size_t oHH = (size_t)l * Hc * Hc, oHF = (size_t)l * Hc * FFc;
    bertcrf_cvt_t<<<gHH, 256, 0, stream>>>(Wq + oHH, wqT + oHH, Hc, Hc);
    bertcrf_cvt_t<<<gHH, 256, 0, stream>>>(Wk + oHH, wkT + oHH, Hc, Hc);
    bertcrf_cvt_t<<<gHH, 256, 0, stream>>>(Wv + oHH, wvT + oHH, Hc, Hc);
    bertcrf_cvt_t<<<gHH, 256, 0, stream>>>(Wo + oHH, woT + oHH, Hc, Hc);
    bertcrf_cvt_t<<<gHF, 256, 0, stream>>>(W1 + oHF, w1T + oHF, Hc, FFc);   // -> [FF][H]
    bertcrf_cvt_t<<<gHF, 256, 0, stream>>>(W2 + oHF, w2T + oHF, FFc, Hc);   // -> [H][FF]
  }

  bertcrf_embed_ln<<<Tc, 256, 0, stream>>>(tok, ttype, we, pe, te, eg, eb, h, hb);

  dim3 gH(Hc / 128, Tc / 128);    // 6 x 64
  dim3 gF(FFc / 128, Tc / 128);   // 24 x 64
  dim3 gA(Bc * NHc, Sc / 32);     // 384 x 8

  for (int l = 0; l < NLc; ++l) {
    size_t oHH = (size_t)l * Hc * Hc, oHF = (size_t)l * Hc * FFc;
    bertcrf_gemm<false, true><<<gH, 256, 0, stream>>>(hb, wqT + oHH, bq + (size_t)l * Hc, qb, Tc, Hc, Hc);
    bertcrf_gemm<false, true><<<gH, 256, 0, stream>>>(hb, wkT + oHH, bk + (size_t)l * Hc, kb, Tc, Hc, Hc);
    bertcrf_gemm<false, true><<<gH, 256, 0, stream>>>(hb, wvT + oHH, bv + (size_t)l * Hc, vb, Tc, Hc, Hc);

    bertcrf_attn<<<gA, 256, 0, stream>>>(qb, kb, vb, amask, ctxb);

    bertcrf_gemm<false, false><<<gH, 256, 0, stream>>>(ctxb, woT + oHH, bo + (size_t)l * Hc, tmp2, Tc, Hc, Hc);
    bertcrf_add_ln<<<Tc, 256, 0, stream>>>(h, tmp2, ln1g + (size_t)l * Hc, ln1b + (size_t)l * Hc, hb);

    bertcrf_gemm<true, true><<<gF, 256, 0, stream>>>(hb, w1T + oHF, fb1 + (size_t)l * FFc, ffb, Tc, FFc, Hc);
    bertcrf_gemm<false, false><<<gH, 256, 0, stream>>>(ffb, w2T + oHF, fb2 + (size_t)l * Hc, tmp2, Tc, Hc, FFc);
    bertcrf_add_ln<<<Tc, 256, 0, stream>>>(h, tmp2, ln2g + (size_t)l * Hc, ln2b + (size_t)l * Hc, hb);
  }

  bertcrf_logits<<<Tc, 256, 0, stream>>>(h, fcw, lgts);
  bertcrf_zero<<<1, 1, 0, stream>>>((float*)d_out);
  bertcrf_crf<<<Bc, 32, 0, stream>>>(lgts, y, amask, cs, ce, ct, (float*)d_out);
}
